// Cat_Linear_Decoder_63230508532406
// MI455X (gfx1250) — matrix operand order repaired and checked by reading, not compiled, not run on hardware
//
#include <hip/hip_runtime.h>
#include <hip/hip_bf16.h>

typedef __attribute__((ext_vector_type(16))) __bf16 v16bf;
typedef __attribute__((ext_vector_type(8)))  float  v8f;

// TDM descriptor group types (probe-verified arities: clang-22 = 5 args, clang-23 = 6 args)
typedef __attribute__((ext_vector_type(4))) unsigned int tdm_v4u;
typedef __attribute__((ext_vector_type(8))) int          tdm_v8i;
typedef __attribute__((ext_vector_type(4))) int          tdm_v4i;

#define N_NODES 100000
#define IN_CH   128
#define HIDDEN  256
#define KDIM    256          // 2*IN_CH
#define E_CNT   500000
#define TWO_E   1000000
#define M_TILE  256          // edges per block
#define LDA     264          // padded row stride (bf16 elems): 528B -> conflict-free b128 column reads
#define THREADS 256
#define W1T_BYTES  (HIDDEN * LDA * 2)          // 135168 B, bf16 image of W1^T incl. padding
#define W1T_QW     (W1T_BYTES / 8)             // 16896 8-byte elements (fits tile_dim0 16 bits)
#define ZB_BYTES   ((size_t)N_NODES * IN_CH * 2)   // 25,600,000 B, bf16 copy of z

union FragU { uint4 q[2]; v16bf v; };

__device__ __forceinline__ unsigned pack_bf16x2(float a, float b) {
  unsigned ua = __builtin_bit_cast(unsigned, a);
  unsigned ub = __builtin_bit_cast(unsigned, b);
  // round-to-nearest-even f32 -> bf16
  ua += 0x7FFFu + ((ua >> 16) & 1u);
  ub += 0x7FFFu + ((ub >> 16) & 1u);
  return (ua >> 16) | (ub & 0xFFFF0000u);
}

// ---- one-time prep: W1 [256x256] f32 -> bf16, transposed + padded, in d_ws ----
// Layout matches the LDS image exactly, so the per-block LDS fill is a flat 1-D TDM copy.
__global__ __launch_bounds__(THREADS)
void prep_w1_kernel(const float* __restrict__ W1, unsigned short* __restrict__ w1t) {
  const int i  = blockIdx.x * THREADS + threadIdx.x;   // 0 .. 32767
  const int kk = i >> 8;         // k-pair 0..127
  const int n  = i & 255;
  const float lo = W1[(2 * kk)     * HIDDEN + n];
  const float hi = W1[(2 * kk + 1) * HIDDEN + n];
  *(unsigned*)(&w1t[n * LDA + 2 * kk]) = pack_bf16x2(lo, hi);
}

// ---- one-time prep: z [100000x128] f32 -> bf16 rows (256 B/node) in d_ws ----
__global__ __launch_bounds__(THREADS)
void prep_z_kernel(const float* __restrict__ z, unsigned short* __restrict__ zb) {
  const size_t i = (size_t)blockIdx.x * THREADS + threadIdx.x;   // float4 index, exact grid
  const float4 v = *(const float4*)(z + i * 4);
  *(uint2*)(zb + i * 4) = make_uint2(pack_bf16x2(v.x, v.y), pack_bf16x2(v.z, v.w));
}

__device__ __forceinline__ void tdm_load_w1(const unsigned short* w1t_ws, void* lds_dst) {
  const unsigned ldsOff = (unsigned)(size_t)lds_dst;            // generic->LDS byte offset (low 32 bits)
  const unsigned long long ga = (unsigned long long)(size_t)w1t_ws;
  // D# group 0: count=1 | lds_addr | global_addr[56:0] | type=2 ("image")
  tdm_v4u g0 = { 1u,
                 ldsOff,
                 (unsigned)ga,
                 (unsigned)((ga >> 32) & 0x1FFFFFFu) | (2u << 30) };
  // D# group 1: data_size=8B (3<<16); tensor_dim0=tile_dim0=W1T_QW; dims 1/2 unused;
  //             tensor_dim0_stride=W1T_QW; no multicast/pad/iterate/barrier.
  tdm_v8i g1 = { (int)(3u << 16),
                 (int)((unsigned)W1T_QW << 16),   // tensor_dim0[15:0] @ bits[63:48]
                 (int)(1u << 16),                 // tensor_dim0[31:16]=0, tensor_dim1=1 @ bits[95:80]
                 (int)((unsigned)W1T_QW << 16),   // tile_dim0 @ bits[127:112]
                 0,                               // tile_dim1=0 (unused), tile_dim2=0
                 (int)W1T_QW,                     // tensor_dim0_stride[31:0]
                 0, 0 };
  tdm_v4i gz = { 0, 0, 0, 0 };
#if defined(__clang_major__) && (__clang_major__ >= 23)
  tdm_v8i gz8 = { 0, 0, 0, 0, 0, 0, 0, 0 };
  __builtin_amdgcn_tensor_load_to_lds(g0, g1, gz, gz, gz8, 0);
#else
  __builtin_amdgcn_tensor_load_to_lds(g0, g1, gz, gz, 0);
#endif
}

// MODE 2: TDM W1 DMA + async bf16 gather (needs full d_ws)
// MODE 1: TDM W1 DMA + f32 gather with VALU pack (needs 132 KB d_ws)
// MODE 0: fully inline staging (no d_ws)
template <int MODE>
__global__ __launch_bounds__(THREADS)
void edge_mlp_kernel(const float* __restrict__ z,
                     const long long* __restrict__ ei,
                     const float* __restrict__ W1,
                     const float* __restrict__ b1,
                     const float* __restrict__ W2,
                     const float* __restrict__ b2,
                     const float* __restrict__ sigb,
                     const unsigned short* __restrict__ w1t_ws,
                     const unsigned short* __restrict__ zb_ws,
                     float* __restrict__ out)
{
  __shared__ unsigned short Alds[M_TILE * LDA];   // 135168 B : gathered rows, bf16
  __shared__ unsigned short W1T [HIDDEN * LDA];   // 135168 B : W1 transposed, bf16
  __shared__ int   sIdx[M_TILE];
  __shared__ int   dIdx[M_TILE];
  __shared__ float b1s [HIDDEN];
  __shared__ float W2s [HIDDEN];
  __shared__ float xLds[M_TILE];

  const int tid = threadIdx.x;
  const long long blockStart = (long long)blockIdx.x * M_TILE;

  // ---- kick off W1^T DMA (wave 0) so it overlaps the whole gather phase ----
  if (MODE >= 1) {
    if (tid < 32) tdm_load_w1(w1t_ws, &W1T[0]);
  }

  // ---- stage edge indices (clamped for the tail block), biases, W2 ----
  {
    long long e = blockStart + tid;
    if (e >= TWO_E) e = TWO_E - 1;
    // symmetrized list: src = ei[e] for both halves; dst = ei[e +/- E]
    sIdx[tid] = (int)ei[e];
    dIdx[tid] = (int)ei[(e < E_CNT) ? (e + E_CNT) : (e - E_CNT)];
    b1s[tid] = b1[tid];
    W2s[tid] = W2[tid];
  }
  __syncthreads();

  // ---- gather rows -> Alds (bf16): A[m][0:128]=z[src], A[m][128:256]=z[dst] ----
  if (MODE == 2) {
    // async 16B-per-lane copies from the pre-converted bf16 node table; no VGPR round trip
    const unsigned ldsBase = (unsigned)(size_t)&Alds[0];
    for (int i = tid; i < M_TILE * 32; i += THREADS) {
      const int m = i >> 5;          // edge row
      const int f = i & 31;          // 16B chunk within 512B bf16 row
      const int node = (f < 16) ? sIdx[m] : dIdx[m];
      const unsigned gOff = ((unsigned)node << 8) + ((unsigned)(f & 15) << 4);
      const unsigned lOff = ldsBase + (unsigned)(m * (LDA * 2) + (f << 4));
      asm volatile("global_load_async_to_lds_b128 %0, %1, %2"
                   :: "v"(lOff), "v"(gOff), "s"(zb_ws) : "memory");
    }
    asm volatile("s_wait_asynccnt 0x0" ::: "memory");
  } else {
    for (int i = tid; i < M_TILE * 64; i += THREADS) {
      const int m = i >> 6;          // edge row
      const int f = i & 63;          // float4 index within 256-wide row
      const int node = (f < 32) ? sIdx[m] : dIdx[m];
      const float4 val = *(const float4*)(z + (size_t)node * IN_CH + ((size_t)(f & 31) << 2));
      const unsigned lo = pack_bf16x2(val.x, val.y);
      const unsigned hi = pack_bf16x2(val.z, val.w);
      *(uint2*)(&Alds[m * LDA + (f << 2)]) = make_uint2(lo, hi);
    }
  }

  if (MODE >= 1) {
    if (tid < 32) __builtin_amdgcn_s_wait_tensorcnt(0);   // wave 0: W1 DMA landed
  } else {
    // fallback: stage W1 transposed from f32 (L2-hot, coalesced reads)
    for (int i = tid; i < (KDIM / 2) * HIDDEN; i += THREADS) {
      const int kk = i >> 8;
      const int n  = i & 255;
      const float lo = W1[(2 * kk)     * HIDDEN + n];
      const float hi = W1[(2 * kk + 1) * HIDDEN + n];
      *(unsigned*)(&W1T[n * LDA + 2 * kk]) = pack_bf16x2(lo, hi);
    }
  }
  __syncthreads();

  // ---- per-wave GEMM: 32 rows x 256 cols, K = 256 ----
  const int lane  = tid & 31;
  const int lm    = lane & 15;     // row (A) / col (B,C) within tile
  const int lg    = lane >> 4;     // lane group selects K sub-range
  const int mBase = (tid >> 5) * 32;

  v8f acc[2][16];
  const v8f vzero = {0.f, 0.f, 0.f, 0.f, 0.f, 0.f, 0.f, 0.f};
  #pragma unroll
  for (int rt = 0; rt < 2; ++rt)
    #pragma unroll
    for (int t = 0; t < 16; ++t) acc[rt][t] = vzero;

  for (int k0 = 0; k0 < KDIM; k0 += 32) {
    // A fragments (16x32 bf16): lane row = lm, elems 0-7 at K+lg*8, elems 8-15 at K+16+lg*8
    v16bf aF[2];
    #pragma unroll
    for (int rt = 0; rt < 2; ++rt) {
      const int row = mBase + rt * 16 + lm;
      FragU u;
      u.q[0] = *(const uint4*)(&Alds[row * LDA + k0 + lg * 8]);
      u.q[1] = *(const uint4*)(&Alds[row * LDA + k0 + 16 + lg * 8]);
      aF[rt] = u.v;
    }
    #pragma unroll
    for (int t = 0; t < 16; ++t) {
      // B fragment (32x16 bf16): lane col = lm, elems 0-7 at K+lg*8, elems 8-15 at K+16+lg*8 (from W1T row)
      const int n = t * 16 + lm;
      FragU ub;
      const uint4* pb = (const uint4*)(&W1T[n * LDA + k0 + lg * 8]);
      ub.q[0] = pb[0];
      ub.q[1] = pb[2];
      acc[0][t] = __builtin_amdgcn_wmma_f32_16x16x32_bf16(
          false, aF[0], false, ub.v, (short)0, acc[0][t], false, false);
      acc[1][t] = __builtin_amdgcn_wmma_f32_16x16x32_bf16(
          false, aF[1], false, ub.v, (short)0, acc[1][t], false, false);
    }
  }

  // ---- fused layer 2: x[m] = sum_n relu(h[m][n] + b1[n]) * W2[n] ----
  float part[16];
  #pragma unroll
  for (int j = 0; j < 16; ++j) part[j] = 0.f;

  #pragma unroll
  for (int t = 0; t < 16; ++t) {
    const int n = t * 16 + lm;
    const float bias = b1s[n];
    const float w2   = W2s[n];
    #pragma unroll
    for (int rt = 0; rt < 2; ++rt)
      #pragma unroll
      for (int j = 0; j < 8; ++j) {
        float h = acc[rt][t][j] + bias;      // C/D layout: row = base + lg*8 + j, col = n
        h = fmaxf(h, 0.f);
        part[rt * 8 + j] = fmaf(h, w2, part[rt * 8 + j]);
      }
  }

  // reduce over the 16 lanes holding the same rows (xor masks < 16 stay in-group)
  #pragma unroll
  for (int msk = 1; msk <= 8; msk <<= 1)
    #pragma unroll
    for (int j = 0; j < 16; ++j)
      part[j] += __shfl_xor(part[j], msk, 32);

  if (lm == 0) {   // lanes 0 and 16 hold disjoint row sets; waves own disjoint 32-row blocks
    #pragma unroll
    for (int rt = 0; rt < 2; ++rt)
      #pragma unroll
      for (int j = 0; j < 8; ++j)
        xLds[mBase + rt * 16 + lg * 8 + j] = part[rt * 8 + j];
  }
  __syncthreads();

  // ---- bias + sigmoid + store ----
  {
    const long long e = blockStart + tid;
    if (e < TWO_E) {
      const float x = xLds[tid] + b2[0] + sigb[0];
      out[e] = 1.0f / (1.0f + __expf(-x));
    }
  }
}

extern "C" void kernel_launch(void* const* d_in, const int* in_sizes, int n_in,
                              void* d_out, int out_size, void* d_ws, size_t ws_size,
                              hipStream_t stream) {
  const float*     z  = (const float*)d_in[0];
  const long long* ei = (const long long*)d_in[1];
  const float*     W1 = (const float*)d_in[2];
  const float*     b1 = (const float*)d_in[3];
  const float*     W2 = (const float*)d_in[4];
  const float*     b2 = (const float*)d_in[5];
  const float*     sb = (const float*)d_in[6];
  float* out = (float*)d_out;

  const int blocks = (TWO_E + M_TILE - 1) / M_TILE;   // 3907
  unsigned short* w1t = (unsigned short*)d_ws;
  unsigned short* zb  = (unsigned short*)((char*)d_ws + W1T_BYTES);  // 256B-aligned offset

  if (ws_size >= (size_t)W1T_BYTES + ZB_BYTES) {
    prep_w1_kernel<<<(KDIM / 2) * HIDDEN / THREADS, THREADS, 0, stream>>>(W1, w1t);
    prep_z_kernel<<<(N_NODES * IN_CH / 4) / THREADS, THREADS, 0, stream>>>(z, zb);  // 12500 blocks
    edge_mlp_kernel<2><<<blocks, THREADS, 0, stream>>>(z, ei, W1, b1, W2, b2, sb, w1t, zb, out);
  } else if (ws_size >= (size_t)W1T_BYTES) {
    prep_w1_kernel<<<(KDIM / 2) * HIDDEN / THREADS, THREADS, 0, stream>>>(W1, w1t);
    edge_mlp_kernel<1><<<blocks, THREADS, 0, stream>>>(z, ei, W1, b1, W2, b2, sb, w1t, nullptr, out);
  } else {
    edge_mlp_kernel<0><<<blocks, THREADS, 0, stream>>>(z, ei, W1, b1, W2, b2, sb, nullptr, nullptr, out);
  }
}
